// Nystroem_31550829756881
// MI455X (gfx1250) — matrix operand order repaired and checked by reading, not compiled, not run on hardware
//
#include <hip/hip_runtime.h>

typedef __attribute__((ext_vector_type(16))) __bf16       v16bf;
typedef __attribute__((ext_vector_type(16))) _Float16     v16h;
typedef __attribute__((ext_vector_type(8)))  float        v8f;
typedef __attribute__((ext_vector_type(8)))  unsigned int v8u;

#define Nn 16384
#define Dd 512
#define Cc 1024

#define TM   32      // rows per block
#define KBLK 64      // staged K-chunk
#define ASTR 520     // halves per A-plane row (512 + 8 pad)
#define DSTR 1032    // halves per dens row   (1024 + 8 pad)
#define BSTR 72      // halves per B-tile row (64 + 8 pad)

// LDS layout (bytes); all fragment bases stay 16B-aligned
#define AHI_OFF   0                         // 32*520*2 = 33280
#define ALO_OFF   33280
#define DEN_OFF   66560                     // 32*1032*2 = 66048 (f16 densities)
#define BHI_OFF   132608                    // 64*72*2 = 9216
#define BLO_OFF   141824
#define X2_OFF    151040                    // 32 floats
#define SMEM_BYTES 151296

__device__ __forceinline__ unsigned short f2bf(float f) {
  unsigned u = __float_as_uint(f);
  u += 0x7FFFu + ((u >> 16) & 1u);          // round-to-nearest-even
  return (unsigned short)(u >> 16);
}
__device__ __forceinline__ float bf2f(unsigned short h) {
  return __uint_as_float(((unsigned)h) << 16);
}

// A-matrix 16-bit fragment: per-lane chunks p[0..7] and p[16..23]
__device__ __forceinline__ v8u ldA16(const unsigned short* p) {
  uint4 a = *(const uint4*)(p);
  uint4 b = *(const uint4*)(p + 16);
  v8u r; r[0]=a.x; r[1]=a.y; r[2]=a.z; r[3]=a.w; r[4]=b.x; r[5]=b.y; r[6]=b.z; r[7]=b.w;
  return r;
}
// B-matrix 16-bit fragment: 16 contiguous halves
__device__ __forceinline__ v8u ldB16(const unsigned short* p) {
  uint4 a = *(const uint4*)(p);
  uint4 b = *(const uint4*)(p + 8);
  v8u r; r[0]=a.x; r[1]=a.y; r[2]=a.z; r[3]=a.w; r[4]=b.x; r[5]=b.y; r[6]=b.z; r[7]=b.w;
  return r;
}

// --- tiny precompute: c2[j] = sum_k centers[j][k]^2 (exact f32) ---
__global__ void nystroem_c2_kernel(const float* __restrict__ centers,
                                   float* __restrict__ c2) {
  int gid  = blockIdx.x * blockDim.x + threadIdx.x;
  int row  = gid >> 5;
  int lane = gid & 31;
  if (row >= Cc) return;
  const float* cr = centers + (size_t)row * Dd;
  float s = 0.f;
  for (int k = lane * 4; k < Dd; k += 32 * 4) {
    float4 v4 = *(const float4*)(cr + k);
    s += v4.x*v4.x + v4.y*v4.y + v4.z*v4.z + v4.w*v4.w;
  }
#pragma unroll
  for (int off = 16; off > 0; off >>= 1) s += __shfl_down(s, off, 32);
  if (lane == 0) c2[row] = s;
}

// --- fused: densities = exp(-g*dist2) (bf16x3 WMMA) then dens @ norm (f16x2 WMMA) ---
__global__ __launch_bounds__(256, 1)
void nystroem_fused(const float* __restrict__ inputs,
                    const float* __restrict__ centers,
                    const float* __restrict__ gammap,
                    const float* __restrict__ norm,
                    const float* __restrict__ c2g,
                    float* __restrict__ outp) {
  extern __shared__ char smem[];
  unsigned short* Ah  = (unsigned short*)(smem + AHI_OFF);
  unsigned short* Al  = (unsigned short*)(smem + ALO_OFF);
  _Float16*       Dn  = (_Float16*)(smem + DEN_OFF);
  unsigned short* Bh  = (unsigned short*)(smem + BHI_OFF);
  unsigned short* Bl  = (unsigned short*)(smem + BLO_OFF);
  float*          x2s = (float*)(smem + X2_OFF);

  const int t    = threadIdx.x;
  const int lane = t & 31, wave = t >> 5;
  const int grp  = lane >> 4, lq = lane & 15;
  const int tm   = wave & 1, tn = wave >> 1;        // 2 x 4 tile grid per block
  const int m0   = blockIdx.x * TM;
  const float gamma = *gammap;

  if (t < TM) x2s[t] = 0.f;
  __syncthreads();

  // Stage input rows -> bf16 hi/lo planes (whole K resident) + exact f32 row norms
  {
    const int r  = t >> 3;
    const int c0 = (t & 7) * 64;
    const float* arow = inputs + (size_t)(m0 + r) * Dd + c0;
    unsigned short* ah = Ah + r * ASTR + c0;
    unsigned short* al = Al + r * ASTR + c0;
    float s = 0.f;
#pragma unroll 4
    for (int i = 0; i < 64; i += 4) {
      float4 v4 = *(const float4*)(arow + i);
      float vv[4] = {v4.x, v4.y, v4.z, v4.w};
#pragma unroll
      for (int j = 0; j < 4; ++j) {
        float v = vv[j];
        s += v * v;
        unsigned short h = f2bf(v);
        ah[i + j] = h;
        al[i + j] = f2bf(v - bf2f(h));
      }
    }
    atomicAdd(&x2s[r], s);
  }
  __syncthreads();

  // ---------------- Phase 1: densities[32][1024] ----------------
  for (int cb = 0; cb < Cc; cb += 64) {
    // three independent accumulators: hi*hi, hi*lo, lo*hi (no intra-step RAW chain)
    v8f acc_hh = {};
    v8f acc_hl = {};
    v8f acc_lh = {};
    for (int kb = 0; kb < Dd; kb += KBLK) {
      __syncthreads();
      for (int i = t; i < 64 * KBLK; i += 256) {       // coalesced along k
        int cl = i >> 6, kl = i & 63;
        float v = centers[(size_t)(cb + cl) * Dd + kb + kl];
        unsigned short h = f2bf(v);
        Bh[cl * BSTR + kl] = h;
        Bl[cl * BSTR + kl] = f2bf(v - bf2f(h));
      }
      __syncthreads();
#pragma unroll
      for (int ks = 0; ks < KBLK; ks += 32) {
        const unsigned short* ap  = Ah + (tm * 16 + lq) * ASTR + (kb + ks) + grp * 16;
        const unsigned short* alp = Al + (tm * 16 + lq) * ASTR + (kb + ks) + grp * 16;
        v16bf afh = __builtin_bit_cast(v16bf, ldB16(ap));
        v16bf afl = __builtin_bit_cast(v16bf, ldB16(alp));
        const unsigned short* bp  = Bh + (tn * 16 + lq) * BSTR + ks + grp * 16;
        const unsigned short* blp = Bl + (tn * 16 + lq) * BSTR + ks + grp * 16;
        v16bf bfh = __builtin_bit_cast(v16bf, ldB16(bp));
        v16bf bfl = __builtin_bit_cast(v16bf, ldB16(blp));
        // bf16x3 split across disjoint accumulators -> WMMAs pipeline back-to-back
        acc_lh = __builtin_amdgcn_wmma_f32_16x16x32_bf16(false, afl, false, bfh, (short)0, acc_lh, false, false);
        acc_hl = __builtin_amdgcn_wmma_f32_16x16x32_bf16(false, afh, false, bfl, (short)0, acc_hl, false, false);
        acc_hh = __builtin_amdgcn_wmma_f32_16x16x32_bf16(false, afh, false, bfh, (short)0, acc_hh, false, false);
      }
    }
    // epilogue: exp(-gamma * max(x2 + c2 - 2*xc, 0)) -> f16 density plane
    float c2v = c2g[cb + tn * 16 + lq];
#pragma unroll
    for (int v = 0; v < 8; ++v) {
      int ml = tm * 16 + grp * 8 + v;
      float xc = acc_hh[v] + (acc_hl[v] + acc_lh[v]);
      float sq = fmaxf(x2s[ml] + c2v - 2.0f * xc, 0.0f);
      Dn[ml * DSTR + (cb + tn * 16 + lq)] = (_Float16)__expf(-gamma * sq);
    }
  }
  __syncthreads();

  // ---------------- Phase 2: out[32][1024] = dens @ norm ----------------
  _Float16* Fh = (_Float16*)(smem + BHI_OFF);
  _Float16* Fl = (_Float16*)(smem + BLO_OFF);
  for (int jb = 0; jb < Cc; jb += 64) {
    v8f acc_h = {};
    v8f acc_l = {};
    for (int kc = 0; kc < Cc; kc += KBLK) {
      __syncthreads();
      if (kc + KBLK < Cc && t < KBLK)                 // warm next norm chunk in L2
        __builtin_prefetch(norm + (size_t)(kc + KBLK + t) * Cc + jb, 0, 1);
      for (int i = t; i < KBLK * 64; i += 256) {      // coalesced along j; transposed store
        int kl = i >> 6, jl = i & 63;
        float v = norm[(size_t)(kc + kl) * Cc + jb + jl];
        _Float16 h = (_Float16)v;
        Fh[jl * BSTR + kl] = h;
        Fl[jl * BSTR + kl] = (_Float16)(v - (float)h);
      }
      __syncthreads();
#pragma unroll
      for (int ks = 0; ks < KBLK; ks += 32) {
        const _Float16* ap  = Dn + (tm * 16 + lq) * DSTR + (kc + ks) + grp * 16;
        v16h af = __builtin_bit_cast(v16h, ldB16((const unsigned short*)ap));
        const _Float16* bp  = Fh + (tn * 16 + lq) * BSTR + ks + grp * 16;
        const _Float16* blp = Fl + (tn * 16 + lq) * BSTR + ks + grp * 16;
        v16h bhv = __builtin_bit_cast(v16h, ldB16((const unsigned short*)bp));
        v16h blv = __builtin_bit_cast(v16h, ldB16((const unsigned short*)blp));
        // f16x2 split across disjoint accumulators
        acc_l = __builtin_amdgcn_wmma_f32_16x16x32_f16(false, af, false, blv, (short)0, acc_l, false, false);
        acc_h = __builtin_amdgcn_wmma_f32_16x16x32_f16(false, af, false, bhv, (short)0, acc_h, false, false);
      }
    }
#pragma unroll
    for (int v = 0; v < 8; ++v) {
      int ml = tm * 16 + grp * 8 + v;
      outp[(size_t)(m0 + ml) * Cc + (jb + tn * 16 + lq)] = acc_h[v] + acc_l[v];
    }
  }
}

extern "C" void kernel_launch(void* const* d_in, const int* in_sizes, int n_in,
                              void* d_out, int out_size, void* d_ws, size_t ws_size,
                              hipStream_t stream) {
  (void)in_sizes; (void)n_in; (void)out_size; (void)ws_size;
  const float* inputs  = (const float*)d_in[0];
  const float* centers = (const float*)d_in[1];
  const float* gammap  = (const float*)d_in[2];
  const float* norm    = (const float*)d_in[3];
  float* outp = (float*)d_out;
  float* c2g  = (float*)d_ws;   // 4 KB scratch only

  static_assert(SMEM_BYTES <= 160 * 1024, "LDS budget");
  (void)hipFuncSetAttribute(reinterpret_cast<const void*>(nystroem_fused),
                            hipFuncAttributeMaxDynamicSharedMemorySize, SMEM_BYTES);

  nystroem_c2_kernel<<<dim3((Cc * 32) / 256), dim3(256), 0, stream>>>(centers, c2g);
  nystroem_fused<<<dim3(Nn / TM), dim3(256), SMEM_BYTES, stream>>>(
      inputs, centers, gammap, norm, c2g, outp);
}
